// Avey_84679575208507
// MI455X (gfx1250) — compile-verified, matrix operand order checked by reading, not run on hardware
//
#include <hip/hip_runtime.h>
#include <hip/hip_bf16.h>

// ---- problem constants (match reference) ----
#define BB   4
#define NN   4096
#define DD   512
#define EE   2048
#define HEADc 1024
#define HALFc 512
#define CATc  1536          // HEAD + HALF
#define BN   (BB*NN)        // 16384 token rows
#define KTS  48             // padded row stride (halfwords) of transposed K block in LDS

typedef __attribute__((ext_vector_type(16))) __bf16 v16bf;
typedef __attribute__((ext_vector_type(8)))  float  v8f;

// round-to-nearest-even fp32 -> bf16 (raw bits)
__device__ __forceinline__ unsigned short f2bf(float f) {
    union { float f; unsigned u; } v; v.f = f;
    unsigned r = v.u + 0x7FFFu + ((v.u >> 16) & 1u);
    return (unsigned short)(r >> 16);
}

__device__ __forceinline__ v8f wmma_bf16(v16bf a, v16bf b, v8f c) {
    return __builtin_amdgcn_wmma_f32_16x16x32_bf16(false, a, false, b,
                                                   (short)0, c, false, false);
}

// ---------------- prep: coalesced tiled transpose + bf16 convert ----------------
// out[n*R + k] = bf16(in[k*C + n]);  R, C multiples of 32. grid (C/32, R/32), block (32,8)
__global__ void k_transpose_bf16(const float* __restrict__ in,
                                 unsigned short* __restrict__ outb,
                                 int R, int C) {
    __shared__ float tile[32][33];
    const int c0 = blockIdx.x * 32, r0 = blockIdx.y * 32;
    for (int i = threadIdx.y; i < 32; i += 8)
        tile[i][threadIdx.x] = in[(size_t)(r0 + i) * C + c0 + threadIdx.x];
    __syncthreads();
    for (int i = threadIdx.y; i < 32; i += 8)
        outb[(size_t)(c0 + i) * R + r0 + threadIdx.x] = f2bf(tile[threadIdx.x][i]);
}

// ---------------- RMSNorm -> bf16 ----------------
__global__ void k_rms(const float* __restrict__ x, const float* __restrict__ g,
                      unsigned short* __restrict__ xnb) {
    __shared__ float red[256];
    const int row = blockIdx.x, tid = threadIdx.x;
    const float* xr = x + (size_t)row * DD;
    float s = 0.f;
    for (int i = tid; i < DD; i += 256) { float v = xr[i]; s += v * v; }
    red[tid] = s; __syncthreads();
    for (int off = 128; off > 0; off >>= 1) {
        if (tid < off) red[tid] += red[tid + off];
        __syncthreads();
    }
    float scale = rsqrtf(red[0] * (1.0f / DD) + 1e-6f);
    for (int i = tid; i < DD; i += 256)
        xnb[(size_t)row * DD + i] = f2bf(xr[i] * scale * g[i]);
}

// ---------------- GEMM1: enriched = relu(xn @ W1 + b1)^2, split ----------------
// grid (BN/128, E/128), 256 threads = 8 waves; wave tile 32(M) x 64(N)
__global__ void k_gemm1(const unsigned short* __restrict__ A,   // xn  BN x D (bf16)
                        const unsigned short* __restrict__ Bt,  // W1t E x D (bf16)
                        const float* __restrict__ b1,
                        unsigned short* __restrict__ cat,       // BN x 1536 (bf16)
                        float* __restrict__ xl,                 // BN x 512
                        float* __restrict__ xr) {               // BN x 512
    const int tid = threadIdx.x, lane = tid & 31, wid = tid >> 5;
    const int l16 = lane & 15, hh = lane >> 4;
    const int m0 = blockIdx.x * 128 + (wid & 3) * 32;
    const int c0 = blockIdx.y * 128 + (wid >> 2) * 64;
    const v8f vz = {};
    v8f acc[2][4];
    for (int i = 0; i < 2; ++i) for (int f = 0; f < 4; ++f) acc[i][f] = vz;

    for (int kk = 0; kk < DD; kk += 32) {
        v16bf a0 = *(const v16bf*)(A + (size_t)(m0 + l16)      * DD + kk + hh * 16);
        v16bf a1 = *(const v16bf*)(A + (size_t)(m0 + 16 + l16) * DD + kk + hh * 16);
#pragma unroll
        for (int f = 0; f < 4; ++f) {
            v16bf b = *(const v16bf*)(Bt + (size_t)(c0 + f * 16 + l16) * DD + kk + hh * 16);
            acc[0][f] = wmma_bf16(a0, b, acc[0][f]);
            acc[1][f] = wmma_bf16(a1, b, acc[1][f]);
        }
    }
    for (int i = 0; i < 2; ++i)
        for (int f = 0; f < 4; ++f) {
            const int col = c0 + f * 16 + l16;
            const float bb = b1[col];
#pragma unroll
            for (int v = 0; v < 8; ++v) {
                const size_t row = m0 + i * 16 + hh * 8 + v;
                float e = acc[i][f][v] + bb;
                e = e > 0.f ? e * e : 0.f;
                if (col < HEADc)                cat[row * CATc + col] = f2bf(e);
                else if (col < HEADc + HALFc)   xl[row * HALFc + (col - HEADc)] = e;
                else                            xr[row * HALFc + (col - HEADc - HALFc)] = e;
            }
        }
}

// ---------------- row L2-normalize x_right -> bf16 ----------------
__global__ void k_rownorm(const float* __restrict__ xr, unsigned short* __restrict__ xrn) {
    __shared__ float red[256];
    const int row = blockIdx.x, tid = threadIdx.x;
    const float* r = xr + (size_t)row * HALFc;
    float s = 0.f;
    for (int i = tid; i < HALFc; i += 256) { float v = r[i]; s += v * v; }
    red[tid] = s; __syncthreads();
    for (int off = 128; off > 0; off >>= 1) {
        if (tid < off) red[tid] += red[tid + off];
        __syncthreads();
    }
    float inv = 1.0f / fmaxf(sqrtf(red[0]), 1e-12f);
    for (int i = tid; i < HALFc; i += 256)
        xrn[(size_t)row * HALFc + i] = f2bf(r[i] * inv);
}

// ---------------- fused context: ctx = sigmoid((V .* (Q K^T)) K + bias), gate ----------------
// grid (N/64, B), 256 threads = 8 waves. Wave (mf=wid&3, ch=wid>>2):
//   S frag:  rows [mf*16,+16), cols [j0+ch*16,+16)     (of the 64x32 S tile)
//   O accum: rows [mf*16,+16), channels [ch*256,+256)  (16 frags)
__global__ void k_ctx(const unsigned short* __restrict__ xrn,   // BN x 512 (bf16)
                      const float* __restrict__ Vm,             // N x N
                      const float* __restrict__ bias,           // 512
                      const float* __restrict__ xl,             // BN x 512
                      unsigned short* __restrict__ cat) {       // BN x 1536
    __shared__ unsigned short kt[HALFc * KTS];  // transposed K block [channel][j], padded, 48 KB
    __shared__ unsigned short pl[64 * 32];      // P tile bf16 [mloc][jloc], 4 KB
    const int tid = threadIdx.x, lane = tid & 31, wid = tid >> 5;
    const int l16 = lane & 15, hh = lane >> 4;
    const int mf = wid & 3, ch = wid >> 2;
    const size_t baseRow = (size_t)blockIdx.y * NN;
    const int mloc0 = blockIdx.x * 64;
    const size_t m0g = baseRow + mloc0 + mf * 16;

    // preload this wave's 16 Q rows as 16 A-fragments (resident across the j loop)
    v16bf aq[16];
#pragma unroll
    for (int kk = 0; kk < 16; ++kk)
        aq[kk] = *(const v16bf*)(xrn + (m0g + l16) * HALFc + kk * 32 + hh * 16);

    const v8f vz = {};
    v8f oacc[16];
#pragma unroll
    for (int f = 0; f < 16; ++f) oacc[f] = vz;

    for (int j0 = 0; j0 < NN; j0 += 32) {
        // cooperative transposed staging of the K block (32 rows x 512 ch):
        // 8 chunks/thread, each = one global b128 (8 bf16) + 8 LDS b16 scatters.
        // q&31 -> row keeps the 32 lanes of a wave on 64 contiguous LDS bytes.
        for (int q = tid; q < (32 * HALFc) / 8; q += 256) {
            const int rr = q & 31;
            const int c8 = (q >> 5) * 8;
            uint4 d = *(const uint4*)(xrn + (baseRow + j0 + rr) * HALFc + c8);
            const unsigned short* el = (const unsigned short*)&d;
#pragma unroll
            for (int t = 0; t < 8; ++t)
                kt[(c8 + t) * KTS + rr] = el[t];
        }
        if (j0 + 32 < NN && tid < 32)   // prefetch next K block rows
            __builtin_prefetch(xrn + (baseRow + j0 + 32 + tid) * HALFc, 0, 1);
        __syncthreads();

        // S = Q K^T (B-frags = K rows, contiguous in global / WGP$)
        v8f s = vz;
#pragma unroll
        for (int kk = 0; kk < 16; ++kk) {
            v16bf b = *(const v16bf*)(xrn + (baseRow + j0 + ch * 16 + l16) * HALFc
                                      + kk * 32 + hh * 16);
            s = wmma_bf16(aq[kk], b, s);
        }

        // P = V .* S  -> bf16 tile in LDS
        {
            const int ncol = j0 + ch * 16 + l16;
#pragma unroll
            for (int v = 0; v < 8; ++v) {
                const int ml = mf * 16 + hh * 8 + v;
                float w = Vm[(size_t)(mloc0 + ml) * NN + ncol] * s[v];
                pl[ml * 32 + ch * 16 + l16] = f2bf(w);
            }
        }
        __syncthreads();

        // O += P @ Kblk  (A-frag = P rows from LDS, B-frags = kt rows, 1 K-step of 32)
        v16bf ap = *(const v16bf*)(pl + (mf * 16 + l16) * 32 + hh * 16);
#pragma unroll
        for (int f = 0; f < 16; ++f) {
            v16bf b = *(const v16bf*)(kt + (size_t)(ch * 256 + f * 16 + l16) * KTS + hh * 16);
            oacc[f] = wmma_bf16(ap, b, oacc[f]);
        }
        __syncthreads();   // protect kt/pl before next iteration
    }

    // epilogue: sigmoid gate, write gated half of concat buffer
#pragma unroll
    for (int f = 0; f < 16; ++f) {
        const int col = ch * 256 + f * 16 + l16;
        const float bb = bias[col];
#pragma unroll
        for (int v = 0; v < 8; ++v) {
            const size_t row = m0g + hh * 8 + v;
            float c = oacc[f][v] + bb;
            float sig = 1.0f / (1.0f + __expf(-c));
            float gv = xl[row * HALFc + col] * sig;
            cat[row * CATc + HEADc + col] = f2bf(gv);
        }
    }
}

// ---------------- final GEMM: out = x + cat @ Wf ----------------
// grid (BN/128, D/128), 256 threads = 8 waves; wave tile 32 x 64
__global__ void k_gemmf(const unsigned short* __restrict__ Acat,  // BN x 1536 (bf16)
                        const unsigned short* __restrict__ Wft,   // D x 1536 (bf16, transposed)
                        const float* __restrict__ x,
                        float* __restrict__ out) {
    const int tid = threadIdx.x, lane = tid & 31, wid = tid >> 5;
    const int l16 = lane & 15, hh = lane >> 4;
    const int m0 = blockIdx.x * 128 + (wid & 3) * 32;
    const int c0 = blockIdx.y * 128 + (wid >> 2) * 64;
    const v8f vz = {};
    v8f acc[2][4];
    for (int i = 0; i < 2; ++i) for (int f = 0; f < 4; ++f) acc[i][f] = vz;

    for (int kk = 0; kk < CATc; kk += 32) {
        v16bf a0 = *(const v16bf*)(Acat + (size_t)(m0 + l16)      * CATc + kk + hh * 16);
        v16bf a1 = *(const v16bf*)(Acat + (size_t)(m0 + 16 + l16) * CATc + kk + hh * 16);
#pragma unroll
        for (int f = 0; f < 4; ++f) {
            v16bf b = *(const v16bf*)(Wft + (size_t)(c0 + f * 16 + l16) * CATc + kk + hh * 16);
            acc[0][f] = wmma_bf16(a0, b, acc[0][f]);
            acc[1][f] = wmma_bf16(a1, b, acc[1][f]);
        }
    }
    for (int i = 0; i < 2; ++i)
        for (int f = 0; f < 4; ++f) {
            const int col = c0 + f * 16 + l16;
#pragma unroll
            for (int v = 0; v < 8; ++v) {
                const size_t row = m0 + i * 16 + hh * 8 + v;
                out[row * DD + col] = x[row * DD + col] + acc[i][f][v];
            }
        }
}

// ---------------- launch ----------------
extern "C" void kernel_launch(void* const* d_in, const int* in_sizes, int n_in,
                              void* d_out, int out_size, void* d_ws, size_t ws_size,
                              hipStream_t stream) {
    const float* x    = (const float*)d_in[0];
    const float* g    = (const float*)d_in[1];
    const float* W1   = (const float*)d_in[2];
    const float* b1   = (const float*)d_in[3];
    const float* Vm   = (const float*)d_in[4];
    const float* bias = (const float*)d_in[5];
    const float* Wf   = (const float*)d_in[6];
    float* out = (float*)d_out;

    char* ws = (char*)d_ws;
    size_t off = 0;
    auto carve = [&](size_t bytes) {
        void* p = ws + off;
        off += (bytes + 255) & ~(size_t)255;
        return p;
    };
    unsigned short* xnb = (unsigned short*)carve((size_t)BN * DD * 2);    // 16 MB
    unsigned short* W1t = (unsigned short*)carve((size_t)EE * DD * 2);    //  2 MB
    unsigned short* Wft = (unsigned short*)carve((size_t)DD * CATc * 2);  // 1.5 MB
    unsigned short* cat = (unsigned short*)carve((size_t)BN * CATc * 2);  // 48 MB
    float*          xl  = (float*)carve((size_t)BN * HALFc * 4);          // 32 MB
    float*          xr  = (float*)carve((size_t)BN * HALFc * 4);          // 32 MB
    unsigned short* xrn = (unsigned short*)carve((size_t)BN * HALFc * 2); // 16 MB

    // W1 (D x E) -> W1t (E x D);  Wf (1536 x 512) -> Wft (512 x 1536)
    k_transpose_bf16<<<dim3(EE / 32, DD / 32), dim3(32, 8), 0, stream>>>(W1, W1t, DD, EE);
    k_transpose_bf16<<<dim3(DD / 32, CATc / 32), dim3(32, 8), 0, stream>>>(Wf, Wft, CATc, DD);
    k_rms<<<dim3(BN), dim3(256), 0, stream>>>(x, g, xnb);
    k_gemm1<<<dim3(BN / 128, EE / 128), dim3(256), 0, stream>>>(xnb, W1t, b1, cat, xl, xr);
    k_rownorm<<<dim3(BN), dim3(256), 0, stream>>>(xr, xrn);
    k_ctx<<<dim3(NN / 64, BB), dim3(256), 0, stream>>>(xrn, Vm, bias, xl, cat);
    k_gemmf<<<dim3(BN / 128, DD / 128), dim3(256), 0, stream>>>(cat, Wft, x, out);
}
